// MoleculeEncoder_53377853554930
// MI455X (gfx1250) — matrix operand order repaired and checked by reading, not compiled, not run on hardware
//
#include <hip/hip_runtime.h>
#include <hip/hip_bf16.h>

// ---------------- problem constants ----------------
#define NNODES 50000
#define NEDGES 200000
#define NGRAPH 2048
#define HDIM   300
#define H2     600
#define NLAYER 5
#define AF     9
#define AV     128
#define BF_    3
#define BV     16

// padded GEMM shapes
#define KP1 320    // K for GEMM1 (300 -> 320)
#define NP1 640    // N-cols for GEMM1 (600 -> 640)
#define KP2 640    // K for GEMM2 (600 -> 640)
#define NP2 320    // N-cols for GEMM2 (300 -> 320)
#define MPAD 50048 // rows padded to multiple of 128 (391 blocks)

typedef __attribute__((ext_vector_type(16))) __bf16 v16bf;
typedef __attribute__((ext_vector_type(8)))  __bf16 v8bf;
typedef __attribute__((ext_vector_type(8)))  float  v8f;

union V16U { v16bf v; v8bf h[2]; unsigned short s[16]; };

__device__ __forceinline__ unsigned short f2bf(float f) {
  unsigned u = __float_as_uint(f);
  u += 0x7fffu + ((u >> 16) & 1u);   // round to nearest even
  return (unsigned short)(u >> 16);
}
__device__ __forceinline__ float bf2f(unsigned short s) {
  return __uint_as_float(((unsigned)s) << 16);
}

// ---------------- weight transpose + pad to bf16 ----------------
// Wt1 [l][n<NP1][k<KP1] <- W1 [l][k<300][n<600]
__global__ void k_w1t(const float* __restrict__ W1, unsigned short* __restrict__ Wt1) {
  int idx = blockIdx.x * blockDim.x + threadIdx.x;
  if (idx >= NLAYER * NP1 * KP1) return;
  int l = idx / (NP1 * KP1);
  int r = idx % (NP1 * KP1);
  int n = r / KP1, kk = r % KP1;
  float v = (n < H2 && kk < HDIM) ? W1[(size_t)l * HDIM * H2 + (size_t)kk * H2 + n] : 0.f;
  Wt1[idx] = f2bf(v);
}
// Wt2 [l][n<NP2][k<KP2] <- W2 [l][k<600][n<300]
__global__ void k_w2t(const float* __restrict__ W2, unsigned short* __restrict__ Wt2) {
  int idx = blockIdx.x * blockDim.x + threadIdx.x;
  if (idx >= NLAYER * NP2 * KP2) return;
  int l = idx / (NP2 * KP2);
  int r = idx % (NP2 * KP2);
  int n = r / KP2, kk = r % KP2;
  float v = (n < HDIM && kk < H2) ? W2[(size_t)l * H2 * HDIM + (size_t)kk * HDIM + n] : 0.f;
  Wt2[idx] = f2bf(v);
}

// ---------------- encoders ----------------
__global__ void k_atom(const int* __restrict__ x, const float* __restrict__ emb,
                       float* __restrict__ h) {
  int n = blockIdx.x;
  int rows[AF];
#pragma unroll
  for (int f = 0; f < AF; ++f) rows[f] = x[n * AF + f] + f * AV;
  for (int c = threadIdx.x; c < HDIM; c += blockDim.x) {
    float s = 0.f;
#pragma unroll
    for (int f = 0; f < AF; ++f) s += emb[(size_t)rows[f] * HDIM + c];
    h[(size_t)n * HDIM + c] = s;
  }
}
__global__ void k_bond(const int* __restrict__ ea, const float* __restrict__ emb,
                       float* __restrict__ e) {
  int n = blockIdx.x;
  int rows[BF_];
#pragma unroll
  for (int f = 0; f < BF_; ++f) rows[f] = ea[n * BF_ + f] + f * BV;
  for (int c = threadIdx.x; c < HDIM; c += blockDim.x) {
    float s = 0.f;
#pragma unroll
    for (int f = 0; f < BF_; ++f) s += emb[(size_t)rows[f] * HDIM + c];
    e[(size_t)n * HDIM + c] = s;
  }
}

// ---------------- per-layer helpers ----------------
__global__ void k_zero_stats(float* __restrict__ stats) {
  int i = blockIdx.x * blockDim.x + threadIdx.x;
  if (i < 1920) stats[i] = 0.f;
}
// Z fp32 [MPAD,KP1] = h (padded with zeros)
__global__ void k_init_z(const float* __restrict__ h, float* __restrict__ Zf) {
  int idx = blockIdx.x * blockDim.x + threadIdx.x;
  if (idx >= MPAD * KP1) return;
  int row = idx / KP1, col = idx % KP1;
  Zf[idx] = (row < NNODES && col < HDIM) ? h[(size_t)row * HDIM + col] : 0.f;
}
// scatter: Z[dst] += relu(h[src] + e)
__global__ void k_edge_scatter(const int* __restrict__ ei, const float* __restrict__ h,
                               const float* __restrict__ e, float* __restrict__ Zf) {
  int eid = blockIdx.x;
  int src = ei[eid];
  int dst = ei[NEDGES + eid];
  const float* hs = h + (size_t)src * HDIM;
  const float* ep = e + (size_t)eid * HDIM;
  float* zp = Zf + (size_t)dst * KP1;
  for (int c = threadIdx.x; c < HDIM; c += blockDim.x) {
    float v = hs[c] + ep[c];
    if (v > 0.f) unsafeAtomicAdd(zp + c, v);
  }
}
// bf16 copy of Z
__global__ void k_z2bf(const float* __restrict__ Zf, unsigned short* __restrict__ Zb) {
  int idx = blockIdx.x * blockDim.x + threadIdx.x;
  if (idx >= MPAD * KP1) return;
  Zb[idx] = f2bf(Zf[idx]);
}

// ---------------- WMMA GEMM (bf16 x bf16 -> fp32 accum) ----------------
// A [MPAD, K] bf16 (row-major), Wt [NP, K] bf16 (i.e. B transposed, row-major).
// out = A @ Wt^T + bias; writes either bf16 outB or fp32 outF; accumulates
// per-column sum / sum-of-squares over valid rows (< NNODES) into sS / sQ.
// Block = 256 thr = 8 waves, wave tile = 16 rows x 64 cols, block = 128 x 64.
__global__ __launch_bounds__(256)
void k_gemm_wmma(const unsigned short* __restrict__ A,
                 const unsigned short* __restrict__ Wt,
                 const float* __restrict__ bias, int biasN,
                 unsigned short* __restrict__ outB,  // bf16 or nullptr
                 float* __restrict__ outF,           // fp32 or nullptr
                 float* __restrict__ sS, float* __restrict__ sQ,
                 int K, int NP) {
  __shared__ float lsum[64], lsq[64];
  const int w    = threadIdx.x >> 5;
  const int lane = threadIdx.x & 31;
  const int m    = lane & 15;        // A-row / C-col within tile
  const int hi   = lane >> 4;
  const int rowBase = blockIdx.x * 128 + w * 16;
  const int colBase = blockIdx.y * 64;
  if (threadIdx.x < 64) { lsum[threadIdx.x] = 0.f; lsq[threadIdx.x] = 0.f; }
  __syncthreads();

  v8f acc[4];
#pragma unroll
  for (int t = 0; t < 4; ++t) {
    int col = colBase + t * 16 + m;
    float b = (col < biasN) ? bias[col] : 0.f;
#pragma unroll
    for (int r = 0; r < 8; ++r) acc[t][r] = b;
  }

  const unsigned short* aptr = A + (size_t)(rowBase + m) * K + 16 * hi;
  for (int k = 0; k < K; k += 32) {
    V16U a;
    a.h[0] = *(const v8bf*)(aptr + k);        // K = k + 16*hi + 0..7
    a.h[1] = *(const v8bf*)(aptr + k + 8);    // K = k + 16*hi + 8..15
#pragma unroll
    for (int t = 0; t < 4; ++t) {
      const unsigned short* bp =
          Wt + (size_t)(colBase + t * 16 + m) * K + k + 16 * hi;
      V16U bfr;
      bfr.v = *(const v16bf*)bp;              // K = k + 16*hi + 0..15
      acc[t] = __builtin_amdgcn_wmma_f32_16x16x32_bf16(
          false, a.v, false, bfr.v, (short)0, acc[t], false, false);
    }
  }

  // store + BN statistics
#pragma unroll
  for (int t = 0; t < 4; ++t) {
    int col = colBase + t * 16 + m;
    float s = 0.f, q = 0.f;
#pragma unroll
    for (int r = 0; r < 8; ++r) {
      int row = rowBase + hi * 8 + r;
      float v = acc[t][r];
      size_t o = (size_t)row * NP + col;
      if (outB) outB[o] = f2bf(v);
      else      outF[o] = v;
      if (row < NNODES) { s += v; q += v * v; }
    }
    s += __shfl_xor(s, 16, 32);
    q += __shfl_xor(q, 16, 32);
    if (hi == 0) {
      atomicAdd(&lsum[t * 16 + m], s);   // ds_add_f32
      atomicAdd(&lsq[t * 16 + m], q);
    }
  }
  __syncthreads();
  if (threadIdx.x < 64) {
    unsafeAtomicAdd(&sS[colBase + threadIdx.x], lsum[threadIdx.x]);
    unsafeAtomicAdd(&sQ[colBase + threadIdx.x], lsq[threadIdx.x]);
  }
}

// BN + ReLU on Y1 (bf16 in, bf16 out), zero pad rows/cols
__global__ void k_bn_relu1(const unsigned short* __restrict__ Y1,
                           const float* __restrict__ stats,  // s1@0, q1@640
                           const float* __restrict__ g, const float* __restrict__ be,
                           unsigned short* __restrict__ Y1a) {
  size_t idx = (size_t)blockIdx.x * blockDim.x + threadIdx.x;
  if (idx >= (size_t)MPAD * NP1) return;
  int col = (int)(idx % NP1);
  size_t row = idx / NP1;
  float out = 0.f;
  if (row < NNODES && col < H2) {
    const float inv = 1.f / (float)NNODES;
    float mean = stats[col] * inv;
    float var  = stats[640 + col] * inv - mean * mean;
    float rs   = rsqrtf(var + 1e-5f);
    float v    = bf2f(Y1[idx]);
    out = fmaxf((v - mean) * rs * g[col] + be[col], 0.f);
  }
  Y1a[idx] = f2bf(out);
}

// final BN (+optional ReLU) on Y2 -> compact h [N, HDIM]
__global__ void k_bn2(const float* __restrict__ Y2,
                      const float* __restrict__ stats,  // s2@1280, q2@1600
                      const float* __restrict__ g, const float* __restrict__ be,
                      float* __restrict__ hout, int do_relu) {
  int idx = blockIdx.x * blockDim.x + threadIdx.x;
  if (idx >= NNODES * HDIM) return;
  int col = idx % HDIM, row = idx / HDIM;
  const float inv = 1.f / (float)NNODES;
  float mean = stats[1280 + col] * inv;
  float var  = stats[1600 + col] * inv - mean * mean;
  float rs   = rsqrtf(var + 1e-5f);
  float v = Y2[(size_t)row * NP2 + col];
  float o = (v - mean) * rs * g[col] + be[col];
  if (do_relu) o = fmaxf(o, 0.f);
  hout[idx] = o;
}

// ---------------- pooling ----------------
__global__ void k_zero_pool(float* __restrict__ pool) {
  int i = blockIdx.x * blockDim.x + threadIdx.x;
  if (i < NGRAPH * HDIM) pool[i] = 0.f;
}
__global__ void k_pool(const int* __restrict__ batch, const float* __restrict__ h,
                       float* __restrict__ pool) {
  int n = blockIdx.x;
  int g = batch[n];
  for (int c = threadIdx.x; c < HDIM; c += blockDim.x)
    unsafeAtomicAdd(&pool[(size_t)g * HDIM + c], h[(size_t)n * HDIM + c]);
}

// ---------------- host launch ----------------
extern "C" void kernel_launch(void* const* d_in, const int* in_sizes, int n_in,
                              void* d_out, int out_size, void* d_ws, size_t ws_size,
                              hipStream_t stream) {
  (void)in_sizes; (void)n_in; (void)out_size; (void)ws_size;
  const int*   batch    = (const int*)d_in[0];
  const int*   x        = (const int*)d_in[1];
  const int*   ei       = (const int*)d_in[2];
  const int*   ea       = (const int*)d_in[3];
  const float* atom_emb = (const float*)d_in[4];
  const float* bond_emb = (const float*)d_in[5];
  const float* W1       = (const float*)d_in[6];
  const float* b1       = (const float*)d_in[7];
  const float* g1       = (const float*)d_in[8];
  const float* be1      = (const float*)d_in[9];
  const float* W2       = (const float*)d_in[10];
  const float* b2       = (const float*)d_in[11];
  const float* g2       = (const float*)d_in[12];
  const float* be2      = (const float*)d_in[13];

  // workspace layout (256B aligned blocks)
  char* base = (char*)d_ws;
  size_t off = 0;
  auto take = [&](size_t bytes) { size_t o = off; off += (bytes + 255) & ~(size_t)255; return o; };
  size_t o_e   = take((size_t)NEDGES * HDIM * 4);        // edge features fp32
  size_t o_h   = take((size_t)NNODES * HDIM * 4);        // node features fp32
  size_t o_zf  = take((size_t)MPAD * KP1 * 4);           // Z fp32 (aliased as Y2 fp32: same size)
  size_t o_zb  = take((size_t)MPAD * KP1 * 2);           // Z bf16
  size_t o_y1  = take((size_t)MPAD * NP1 * 2);           // Y1 bf16 (pre-BN)
  size_t o_y1a = take((size_t)MPAD * NP1 * 2);           // Y1 bf16 (post-BN/ReLU)
  size_t o_wt1 = take((size_t)NLAYER * NP1 * KP1 * 2);   // W1^T bf16 padded
  size_t o_wt2 = take((size_t)NLAYER * NP2 * KP2 * 2);   // W2^T bf16 padded
  size_t o_st  = take((size_t)1920 * 4);                 // s1[640] q1[640] s2[320] q2[320]

  float*          ebuf  = (float*)(base + o_e);
  float*          hbuf  = (float*)(base + o_h);
  float*          Zf    = (float*)(base + o_zf);
  float*          Y2f   = Zf;                            // alias: Z dead after GEMM1
  unsigned short* Zb    = (unsigned short*)(base + o_zb);
  unsigned short* Y1    = (unsigned short*)(base + o_y1);
  unsigned short* Y1a   = (unsigned short*)(base + o_y1a);
  unsigned short* Wt1   = (unsigned short*)(base + o_wt1);
  unsigned short* Wt2   = (unsigned short*)(base + o_wt2);
  float*          stats = (float*)(base + o_st);

  float* out_h    = (float*)d_out;                       // [N, HDIM]
  float* out_pool = (float*)d_out + (size_t)NNODES * HDIM; // [G, HDIM]

  // weights -> transposed padded bf16 (once per launch)
  k_w1t<<<(NLAYER * NP1 * KP1 + 255) / 256, 256, 0, stream>>>(W1, Wt1);
  k_w2t<<<(NLAYER * NP2 * KP2 + 255) / 256, 256, 0, stream>>>(W2, Wt2);

  // encoders
  k_atom<<<NNODES, 256, 0, stream>>>(x, atom_emb, hbuf);
  k_bond<<<NEDGES, 256, 0, stream>>>(ea, bond_emb, ebuf);

  const dim3 g1grid(MPAD / 128, NP1 / 64);  // 391 x 10
  const dim3 g2grid(MPAD / 128, NP2 / 64);  // 391 x 5

  for (int i = 0; i < NLAYER; ++i) {
    k_zero_stats<<<8, 256, 0, stream>>>(stats);
    k_init_z<<<(MPAD * KP1 + 255) / 256, 256, 0, stream>>>(hbuf, Zf);
    k_edge_scatter<<<NEDGES, 256, 0, stream>>>(ei, hbuf, ebuf, Zf);
    k_z2bf<<<(MPAD * KP1 + 255) / 256, 256, 0, stream>>>(Zf, Zb);

    k_gemm_wmma<<<g1grid, 256, 0, stream>>>(
        Zb, Wt1 + (size_t)i * NP1 * KP1, b1 + (size_t)i * H2, H2,
        Y1, nullptr, stats + 0, stats + 640, KP1, NP1);

    k_bn_relu1<<<(int)(((size_t)MPAD * NP1 + 255) / 256), 256, 0, stream>>>(
        Y1, stats, g1 + (size_t)i * H2, be1 + (size_t)i * H2, Y1a);

    k_gemm_wmma<<<g2grid, 256, 0, stream>>>(
        Y1a, Wt2 + (size_t)i * NP2 * KP2, b2 + (size_t)i * HDIM, HDIM,
        nullptr, Y2f, stats + 1280, stats + 1600, KP2, NP2);

    float* dest = (i == NLAYER - 1) ? out_h : hbuf;
    k_bn2<<<(NNODES * HDIM + 255) / 256, 256, 0, stream>>>(
        Y2f, stats, g2 + (size_t)i * HDIM, be2 + (size_t)i * HDIM,
        dest, (i < NLAYER - 1) ? 1 : 0);
  }

  // global_add_pool on final h (already in d_out)
  k_zero_pool<<<(NGRAPH * HDIM + 255) / 256, 256, 0, stream>>>(out_pool);
  k_pool<<<NNODES, 256, 0, stream>>>(batch, out_h, out_pool);
}
